// RNN_35759897707259
// MI455X (gfx1250) — matrix operand order repaired and checked by reading, not compiled, not run on hardware
//
#include <hip/hip_runtime.h>
#include <hip/hip_bf16.h>

// Problem sizes (match reference)
#define B_   256
#define T_   256
#define DIN  512
#define DH   512
#define DOUT 512
#define DK   1024   // DIN + DH

typedef __bf16 bf16_t;
typedef __attribute__((ext_vector_type(4)))  __bf16 v4bf;
typedef __attribute__((ext_vector_type(8)))  __bf16 v8bf;
typedef __attribute__((ext_vector_type(16))) __bf16 v16bf;
typedef __attribute__((ext_vector_type(4)))  float  v4f;
typedef __attribute__((ext_vector_type(8)))  float  v8f;

typedef __attribute__((ext_vector_type(4))) unsigned int u32x4;
typedef __attribute__((ext_vector_type(4))) int          i32x4;
typedef __attribute__((ext_vector_type(8))) int          i32x8;

// --- CDNA5 TDM availability -------------------------------------------------
#if defined(__has_builtin)
#if __has_builtin(__builtin_amdgcn_tensor_load_to_lds)
#define HAVE_TDM 1
#endif
#if __has_builtin(__builtin_amdgcn_s_wait_tensorcnt)
#define HAVE_WAIT_TENSOR 1
#endif
#endif
#ifndef HAVE_TDM
#define HAVE_TDM 0
#endif
#ifndef HAVE_WAIT_TENSOR
#define HAVE_WAIT_TENSOR 0
#endif

__device__ __forceinline__ void wait_tensorcnt0() {
#if HAVE_WAIT_TENSOR
  __builtin_amdgcn_s_wait_tensorcnt(0);
#elif HAVE_TDM
  asm volatile("s_wait_tensorcnt 0x0" ::: "memory");
#endif
}

// TDM 2D bf16 tile load: global[tile_h rows x tile_w elems, row stride
// row_stride_elems] -> LDS at lds_byte_addr, with optional LDS row padding
// (pad_dwords inserted every 2^(pad_interval_code+1) DWORDs).
// D# packing per CDNA5 ISA.
__device__ __forceinline__ void tdm_load_2d_bf16(
    unsigned lds_byte_addr, const bf16_t* gptr, unsigned tile_w,
    unsigned tile_h, unsigned long long row_stride_elems, unsigned pad_dwords,
    unsigned pad_interval_code) {
#if HAVE_TDM
  unsigned long long ga = (unsigned long long)(size_t)gptr;
  u32x4 g0;
  g0[0] = 1u;                                           // count=1, user D#
  g0[1] = lds_byte_addr;                                // lds_addr
  g0[2] = (unsigned)ga;                                 // global_addr[31:0]
  g0[3] = (unsigned)((ga >> 32) & 0x01ffffffu) | (2u << 30);  // [56:32]|type=2
  unsigned d0 = (1u << 16);                             // data_size = 2 bytes
  if (pad_dwords) {
    d0 |= (1u << 20);                                   // pad_enable
    d0 |= (pad_interval_code & 7u) << 22;               // pad_interval
    d0 |= ((pad_dwords - 1u) & 0x7fu) << 25;            // pad_amount
  }
  i32x8 g1;
  g1[0] = (int)d0;                                      // mask=0 (no cluster)
  g1[1] = (int)((tile_w & 0xffffu) << 16);              // tensor_dim0[15:0]
  g1[2] = (int)(((tile_w >> 16) & 0xffffu) | ((tile_h & 0xffffu) << 16));
  g1[3] = (int)((tile_w & 0xffffu) << 16);              // tile_dim0
  g1[4] = (int)(tile_h & 0xffffu);                      // tile_dim1 (dim2=0)
  g1[5] = (int)(unsigned)(row_stride_elems & 0xffffffffull);
  g1[6] = (int)(unsigned)((row_stride_elems >> 32) & 0xffffull);
  g1[7] = 0;
  i32x4 z4 = {0, 0, 0, 0};
#if __clang_major__ >= 23
  i32x8 z8 = {0, 0, 0, 0, 0, 0, 0, 0};
  __builtin_amdgcn_tensor_load_to_lds(g0, g1, z4, z4, z8, 0);
#else
  __builtin_amdgcn_tensor_load_to_lds(g0, g1, z4, z4, 0);
#endif
#else
  (void)lds_byte_addr; (void)gptr; (void)tile_w; (void)tile_h;
  (void)row_stride_elems; (void)pad_dwords; (void)pad_interval_code;
#endif
}

// ---------------------------------------------------------------------------
// Conversion kernels: f32 -> bf16 (x), and weight transpose [K][N] -> [N][K]
// so WMMA B-fragments become contiguous 32-byte loads.
// ---------------------------------------------------------------------------
__global__ void cvt_x_kernel(const float* __restrict__ x,
                             bf16_t* __restrict__ xb, int n4) {
  int i = blockIdx.x * blockDim.x + threadIdx.x;
  if (i >= n4) return;
  v4f f = ((const v4f*)x)[i];
  ((v4bf*)xb)[i] = __builtin_convertvector(f, v4bf);
}

__global__ void cvt_wT_kernel(const float* __restrict__ W,   // [K][N]
                              bf16_t* __restrict__ WT,       // [N][K]
                              int N, int K) {
  int i = blockIdx.x * blockDim.x + threadIdx.x;
  if (i >= N * K) return;
  int n = i / K, k = i % K;
  WT[i] = (bf16_t)W[k * N + n];
}

// ---------------------------------------------------------------------------
// Fragment helpers (wave32 WMMA bf16; contiguous K order k=16h+s used by this code, not the ISA table order)
// A 16x32 bf16: lane holds row (lane&15); lanes 0-15: K=0..15,
//               lanes 16-31: K=16..31  -> two 16B chunks.
// B 32x16 bf16 from W^T ([N][K]): lane holds col (lane&15); lanes 0-15:
//               K=0..15, lanes 16-31: K=16..31 -> one 32B chunk.
// ---------------------------------------------------------------------------
__device__ inline v16bf load_a_frag(const bf16_t* __restrict__ rowptr, int kA) {
  v8bf lo = *(const v8bf*)(rowptr + kA);
  v8bf hi = *(const v8bf*)(rowptr + kA + 8);
  return __builtin_shufflevector(lo, hi, 0, 1, 2, 3, 4, 5, 6, 7,
                                 8, 9, 10, 11, 12, 13, 14, 15);
}

// ---------------------------------------------------------------------------
// Phase 1: sequential recurrence. grid = B/16 blocks, 256 threads (8 waves).
// Each block owns 16 batch rows; h lives in LDS across all T steps. x_t tiles
// are staged by the TDM (double-buffered, overlapped with WMMA compute).
// Each wave computes 4 N-tiles (8 waves x 4 x 16 = 512 = DH).
// ---------------------------------------------------------------------------
__global__ __launch_bounds__(256) void rnn_recur_kernel(
    const bf16_t* __restrict__ xb,      // [B,T,DIN] bf16
    const float*  __restrict__ h0,      // [B,DH] f32
    const bf16_t* __restrict__ WhT,     // [DH][DK] bf16 (W_h transposed)
    const float*  __restrict__ bh,      // [DH]
    bf16_t*       __restrict__ h_all,   // [B,T,DH] bf16 (workspace)
    float*        __restrict__ h_final, // [B,DH] f32 (d_out tail)
    const int*    __restrict__ Np) {
  __shared__ __align__(32) bf16_t xs[2][16 * 520];  // double-buffered x tile
  __shared__ __align__(32) bf16_t hs[16 * 520];     // h_t tile

  const int Nsteps = *Np;
  const int b0   = blockIdx.x * 16;
  const int tid  = threadIdx.x;
  const int lane = tid & 31;
  const int wave = tid >> 5;
  const int l15  = lane & 15;        // A-row / B-col / C-col index
  const int hiH  = lane >> 4;        // lane half
  const int kA   = hiH ? 16 : 0;     // A-frag K base within 32-chunk
  const int kB   = hiH ? 16 : 0;     // B-frag K base within 32-chunk

  // init h from h_state
  for (int c = tid; c < 16 * DH; c += 256) {
    int r = c >> 9, k = c & 511;
    hs[r * 520 + k] = (bf16_t)h0[(size_t)(b0 + r) * DH + k];
  }

#if HAVE_TDM
  // prologue: async stage x_0 (rows: batch rows, stride T*DIN elems;
  // LDS pad: 4 DWORDs every 256 DWORDs -> row stride 520 elems)
  if (wave == 0 && Nsteps > 0) {
    tdm_load_2d_bf16((unsigned)(size_t)&xs[0][0],
                     &xb[(size_t)b0 * T_ * DIN], DIN, 16,
                     (unsigned long long)T_ * DIN, 4u, 7u);
  }
#endif
  __syncthreads();

  for (int t = 0; t < Nsteps; ++t) {
    const int buf = t & 1;
    // (a) publish h_t to workspace (16B chunks)
    for (int c = tid; c < 16 * 64; c += 256) {
      int r  = c >> 6;
      int kc = (c & 63) * 8;
      *(v8bf*)&h_all[(size_t)(b0 + r) * T_ * DH + (size_t)t * DH + kc] =
          *(const v8bf*)&hs[r * 520 + kc];
    }
#if HAVE_TDM
    if (wave == 0) wait_tensorcnt0();   // x_t tile landed in xs[buf]
    __syncthreads();
    if (wave == 0 && t + 1 < Nsteps) {  // overlap x_{t+1} DMA with compute
      tdm_load_2d_bf16((unsigned)(size_t)&xs[buf ^ 1][0],
                       &xb[(size_t)b0 * T_ * DIN + (size_t)(t + 1) * DIN],
                       DIN, 16, (unsigned long long)T_ * DIN, 4u, 7u);
    }
#else
    for (int c = tid; c < 16 * 64; c += 256) {
      int r  = c >> 6;
      int kc = (c & 63) * 8;
      *(v8bf*)&xs[buf][r * 520 + kc] =
          *(const v8bf*)&xb[(size_t)(b0 + r) * T_ * DIN + (size_t)t * DIN + kc];
    }
    __syncthreads();
#endif

    // (b) h_new = [x_t, h_t] @ W_h  (K = 1024 in 32-wide WMMA steps)
    v8f acc[4] = {};
    for (int ks = 0; ks < DK / 32; ++ks) {
      const int kbase = ks * 32;
      const bf16_t* src = (kbase < DIN) ? &xs[buf][l15 * 520 + kbase]
                                        : &hs[l15 * 520 + (kbase - DIN)];
      v16bf a = load_a_frag(src, kA);
#pragma unroll
      for (int j = 0; j < 4; ++j) {
        const int n = (wave * 4 + j) * 16 + l15;
        v16bf b = *(const v16bf*)&WhT[(size_t)n * DK + kbase + kB];
        acc[j] = __builtin_amdgcn_wmma_f32_16x16x32_bf16(
            false, a, false, b, (short)0, acc[j], false, false);
      }
    }
    __syncthreads();  // all hs reads complete before overwrite

    // (c) bias + writeback h_new into LDS (C/D layout: VGPR r -> row r/r+8)
#pragma unroll
    for (int j = 0; j < 4; ++j) {
      const int n = (wave * 4 + j) * 16 + l15;
      const float bias = bh[n];
#pragma unroll
      for (int r = 0; r < 8; ++r) {
        const int m = r + (hiH ? 8 : 0);
        hs[m * 520 + n] = (bf16_t)(acc[j][r] + bias);
      }
    }
    __syncthreads();
  }

  // h_final (f32)
  for (int c = tid; c < 16 * DH; c += 256) {
    int r = c >> 9, k = c & 511;
    h_final[(size_t)(b0 + r) * DH + k] = (float)hs[r * 520 + k];
  }
}

// ---------------------------------------------------------------------------
// Phase 2: fully parallel output GEMM + sigmoid over all B*T rows.
// grid = (B*T)/16 blocks; TDM stages the x/h halves of the combined tile
// (rows are contiguous here: row stride = DIN/DH elems).
// 8 waves x 4 N-tiles cover DOUT=512.
// ---------------------------------------------------------------------------
__global__ __launch_bounds__(256) void rnn_out_kernel(
    const bf16_t* __restrict__ xb,     // [(B*T), DIN] bf16
    const bf16_t* __restrict__ h_all,  // [(B*T), DH] bf16
    const bf16_t* __restrict__ WoT,    // [DOUT][DK] bf16
    const float*  __restrict__ bo,     // [DOUT]
    float*        __restrict__ out,    // [(B*T), DOUT] f32
    const int*    __restrict__ Np) {
  __shared__ __align__(32) bf16_t xs2[16 * 520];
  __shared__ __align__(32) bf16_t hs2[16 * 520];

  const int Nsteps = *Np;
  const int m0   = blockIdx.x * 16;
  const int tid  = threadIdx.x;
  const int lane = tid & 31;
  const int wave = tid >> 5;
  const int l15  = lane & 15;
  const int hiH  = lane >> 4;
  const int kA   = hiH ? 16 : 0;
  const int kB   = hiH ? 16 : 0;

#if HAVE_TDM
  if (wave == 0) {
    tdm_load_2d_bf16((unsigned)(size_t)&xs2[0], &xb[(size_t)m0 * DIN],
                     DIN, 16, (unsigned long long)DIN, 4u, 7u);
    tdm_load_2d_bf16((unsigned)(size_t)&hs2[0], &h_all[(size_t)m0 * DH],
                     DH, 16, (unsigned long long)DH, 4u, 7u);
    wait_tensorcnt0();
  }
  __syncthreads();
#else
  for (int c = tid; c < 16 * 128; c += 256) {
    int r  = c >> 7;
    int kc = (c & 127) * 8;
    if (kc < DIN)
      *(v8bf*)&xs2[r * 520 + kc] = *(const v8bf*)&xb[(size_t)(m0 + r) * DIN + kc];
    else
      *(v8bf*)&hs2[r * 520 + (kc - DIN)] =
          *(const v8bf*)&h_all[(size_t)(m0 + r) * DH + (kc - DIN)];
  }
  __syncthreads();
#endif

  v8f acc[4] = {};
  for (int ks = 0; ks < DK / 32; ++ks) {
    const int kbase = ks * 32;
    const bf16_t* src = (kbase < DIN) ? &xs2[l15 * 520 + kbase]
                                      : &hs2[l15 * 520 + (kbase - DIN)];
    v16bf a = load_a_frag(src, kA);
#pragma unroll
    for (int j = 0; j < 4; ++j) {
      const int n = (wave * 4 + j) * 16 + l15;
      __builtin_prefetch(&WoT[(size_t)n * DK + kbase + 512], 0, 1);
      v16bf b = *(const v16bf*)&WoT[(size_t)n * DK + kbase + kB];
      acc[j] = __builtin_amdgcn_wmma_f32_16x16x32_bf16(
          false, a, false, b, (short)0, acc[j], false, false);
    }
  }

  // bias + sigmoid + store
#pragma unroll
  for (int j = 0; j < 4; ++j) {
    const int n = (wave * 4 + j) * 16 + l15;
    const float bias = bo[n];
#pragma unroll
    for (int r = 0; r < 8; ++r) {
      const int m = m0 + r + (hiH ? 8 : 0);
      const int t = m % T_;
      if (t < Nsteps) {
        const float v = acc[j][r] + bias;
        out[(size_t)m * DOUT + n] = 1.0f / (1.0f + __expf(-v));
      }
    }
  }
}

// ---------------------------------------------------------------------------
// Launch
// ---------------------------------------------------------------------------
extern "C" void kernel_launch(void* const* d_in, const int* in_sizes, int n_in,
                              void* d_out, int out_size, void* d_ws,
                              size_t ws_size, hipStream_t stream) {
  const float* x   = (const float*)d_in[0];
  const float* h0  = (const float*)d_in[1];
  const float* W_h = (const float*)d_in[2];
  const float* b_h = (const float*)d_in[3];
  const float* W_o = (const float*)d_in[4];
  const float* b_o = (const float*)d_in[5];
  const int*   Np  = (const int*)d_in[6];

  float* out     = (float*)d_out;                       // [B,T,DOUT]
  float* h_final = out + (size_t)B_ * T_ * DOUT;        // [B,DH]

  // workspace layout
  char* ws = (char*)d_ws;
  bf16_t* WhT  = (bf16_t*)ws;                                   // 1 MB
  bf16_t* WoT  = (bf16_t*)(ws + (1u << 20));                    // 1 MB
  bf16_t* xbf  = (bf16_t*)(ws + (2u << 20));                    // 64 MB
  bf16_t* hall = (bf16_t*)(ws + (2u << 20) +
                           (size_t)B_ * T_ * DIN * sizeof(bf16_t));  // 64 MB

  // conversions
  {
    int n4 = (B_ * T_ * DIN) / 4;
    cvt_x_kernel<<<(n4 + 255) / 256, 256, 0, stream>>>(x, xbf, n4);
    cvt_wT_kernel<<<(DH * DK + 255) / 256, 256, 0, stream>>>(W_h, WhT, DH, DK);
    cvt_wT_kernel<<<(DOUT * DK + 255) / 256, 256, 0, stream>>>(W_o, WoT, DOUT, DK);
  }

  // phase 1: sequential recurrence (batch-parallel only)
  rnn_recur_kernel<<<B_ / 16, 256, 0, stream>>>(xbf, h0, WhT, b_h, hall,
                                                h_final, Np);

  // phase 2: fully parallel output GEMM + sigmoid
  rnn_out_kernel<<<(B_ * T_) / 16, 256, 0, stream>>>(xbf, hall, WoT, b_o, out,
                                                     Np);
}
